// IAU_78176994721951
// MI455X (gfx1250) — matrix operand order repaired and checked by reading, not compiled, not run on hardware
//
#include <hip/hip_runtime.h>
#include <hip/hip_bf16.h>
#include <math.h>

typedef __attribute__((ext_vector_type(8)))  unsigned short ushort8;
typedef __attribute__((ext_vector_type(16))) __bf16        bf16x16;
typedef __attribute__((ext_vector_type(8)))  float         floatx8;

#define B_   32
#define C_   2048
#define HW_  192
#define CI_  1024
#define EPS_ 1e-5f

static __device__ __forceinline__ unsigned short f2bf(float f) {
  unsigned int u = __float_as_uint(f);
  u += 0x7FFFu + ((u >> 16) & 1u);   // round-to-nearest-even
  return (unsigned short)(u >> 16);
}

// Build a 16x32 A-frag / 32x16 B-frag (16 bf16 per lane) from two 16B LDS chunks.
static __device__ __forceinline__ bf16x16 frag_ld(const unsigned short* p0,
                                                  const unsigned short* p1) {
  union { ushort8 h[2]; bf16x16 v; } u;
  u.h[0] = *reinterpret_cast<const ushort8*>(p0);
  u.h[1] = *reinterpret_cast<const ushort8*>(p1);
  return u.v;
}

static __device__ __forceinline__ floatx8 wmma_bf16(bf16x16 a, bf16x16 b, floatx8 c) {
  return __builtin_amdgcn_wmma_f32_16x16x32_bf16(false, a, false, b, (short)0, c,
                                                 false, false);
}

// ---------------------------------------------------------------------------
// K0: f32 -> bf16 conversion (x and w1)
// ---------------------------------------------------------------------------
__global__ void k_f32_to_bf16(const float* __restrict__ in,
                              unsigned short* __restrict__ out, int n) {
  int i = blockIdx.x * blockDim.x + threadIdx.x;
  if (i < n) out[i] = f2bf(in[i]);
}

// ---------------------------------------------------------------------------
// K1: fused Gram-matrix attention (flash style).
//   y[b,m,:] = softmax_d( gx[b,m,:].gx[b,d,:] ) @ gx[b,d,:]
// Block = 128 threads (4 waves); each wave owns a 16-row query strip.
// ---------------------------------------------------------------------------
__global__ void __launch_bounds__(128)
k_flash_gram(const unsigned short* __restrict__ Xb, unsigned short* __restrict__ Yb) {
  __shared__ alignas(16) unsigned short Qs[64][HW_];   // 24 KB  query strip (bf16)
  __shared__ alignas(16) unsigned short Ks[32][HW_];   // 12 KB  key block [d][k]
  __shared__ alignas(16) unsigned short KsT[HW_][32];  // 12 KB  value block [k][d]
  __shared__ alignas(16) unsigned short Ps[4][16][32]; //  4 KB  per-wave P tile

  const int b = blockIdx.y;
  const int m0 = blockIdx.x * 64;
  const int tid = threadIdx.x;
  const int w = tid >> 5;
  const int lane = tid & 31;
  const int l15 = lane & 15;
  const int kh = lane >> 4;

  const unsigned short* Xbb = Xb + (size_t)b * C_ * HW_;

  for (int t = tid; t < 64 * HW_; t += 128) {
    int i = t / HW_, f = t % HW_;
    Qs[i][f] = Xbb[(size_t)(m0 + i) * HW_ + f];
  }
  __syncthreads();

  // Q A-fragments: row = lane&15 of the wave's strip, element->K per ISA layout.
  bf16x16 qf[6];
#pragma unroll
  for (int kk = 0; kk < 6; ++kk)
    qf[kk] = frag_ld(&Qs[w * 16 + l15][kk * 32 + kh * 8],
                     &Qs[w * 16 + l15][kk * 32 + 16 + kh * 8]);

  floatx8 acc[12];
#pragma unroll
  for (int nt = 0; nt < 12; ++nt) acc[nt] = {};
  float rm[8], rl[8];
#pragma unroll
  for (int r = 0; r < 8; ++r) { rm[r] = -1e30f; rl[r] = 0.0f; }

  for (int j = 0; j < C_; j += 32) {
    __syncthreads();
    for (int t = tid; t < 32 * HW_; t += 128) {
      int d = t / HW_, f = t % HW_;
      unsigned short v = Xbb[(size_t)(j + d) * HW_ + f];
      Ks[d][f] = v;    // for S = Q.K^T (B-frag wants contiguous k at fixed d)
      KsT[f][d] = v;   // for O += P.V  (B-frag wants contiguous d at fixed k)
    }
    __syncthreads();
    if (j + 32 < C_) {
      const char* nx = (const char*)(Xbb + (size_t)(j + 32) * HW_);
      __builtin_prefetch(nx + tid * 96, 0, 0);
    }

    // S tiles: 16 rows x 32 keys, K=192 (6 wmma steps per 16-col tile)
    floatx8 s0 = {}, s1 = {};
#pragma unroll
    for (int kk = 0; kk < 6; ++kk) {
      bf16x16 b0 = frag_ld(&Ks[l15][kk * 32 + kh * 8],
                           &Ks[l15][kk * 32 + 16 + kh * 8]);
      bf16x16 b1 = frag_ld(&Ks[16 + l15][kk * 32 + kh * 8],
                           &Ks[16 + l15][kk * 32 + 16 + kh * 8]);
      s0 = wmma_bf16(qf[kk], b0, s0);
      s1 = wmma_bf16(qf[kk], b1, s1);
    }

    // Online softmax. C-layout: vgpr r <-> row r+8*kh, col = l15 (+16 for s1).
    float sc[8];
#pragma unroll
    for (int r = 0; r < 8; ++r) {
      float v0 = s0[r], v1 = s1[r];
      float mx = fmaxf(v0, v1);
#pragma unroll
      for (int off = 1; off < 16; off <<= 1)
        mx = fmaxf(mx, __shfl_xor(mx, off, 32));
      float mnew = fmaxf(rm[r], mx);
      float p0 = __expf(v0 - mnew);
      float p1 = __expf(v1 - mnew);
      float scale = __expf(rm[r] - mnew);
      float ps = p0 + p1;
#pragma unroll
      for (int off = 1; off < 16; off <<= 1)
        ps += __shfl_xor(ps, off, 32);
      rl[r] = rl[r] * scale + ps;
      rm[r] = mnew;
      sc[r] = scale;
      Ps[w][r + 8 * kh][l15] = f2bf(p0);
      Ps[w][r + 8 * kh][16 + l15] = f2bf(p1);
    }

#pragma unroll
    for (int nt = 0; nt < 12; ++nt)
#pragma unroll
      for (int r = 0; r < 8; ++r) acc[nt][r] *= sc[r];

    // P tile round-trips through per-wave LDS to become an A-fragment.
    asm volatile("s_wait_dscnt 0" ::: "memory");
    bf16x16 pf = frag_ld(&Ps[w][l15][kh * 8], &Ps[w][l15][16 + kh * 8]);

#pragma unroll
    for (int nt = 0; nt < 12; ++nt) {
      bf16x16 bv = frag_ld(&KsT[nt * 16 + l15][kh * 8],
                           &KsT[nt * 16 + l15][16 + kh * 8]);
      acc[nt] = wmma_bf16(pf, bv, acc[nt]);
    }
  }

  unsigned short* Ybb = Yb + (size_t)b * C_ * HW_;
#pragma unroll
  for (int nt = 0; nt < 12; ++nt)
#pragma unroll
    for (int r = 0; r < 8; ++r) {
      int row = m0 + w * 16 + r + 8 * kh;
      int col = nt * 16 + l15;
      Ybb[(size_t)row * HW_ + col] = f2bf(acc[nt][r] * (1.0f / rl[r]));
    }
}

// ---------------------------------------------------------------------------
// K2: bf16 WMMA GEMM  Out[b] = A(MxK) * B[b](KxN) + bias   (conv1x1 w1)
// 64x64 tile per block (4 waves), K stepped by 32.
// ---------------------------------------------------------------------------
__global__ void __launch_bounds__(128)
k_gemm_bias(const unsigned short* __restrict__ A, const unsigned short* __restrict__ Bm,
            const float* __restrict__ bias, float* __restrict__ Out,
            int M, int K, int N) {
  __shared__ alignas(16) unsigned short As[64][32];
  __shared__ alignas(16) unsigned short BsT[64][32];  // BsT[n][k] = B[k][n]
  const int b = blockIdx.z;
  const int m0 = blockIdx.x * 64;
  const int n0 = blockIdx.y * 64;
  const int tid = threadIdx.x;
  const int w = tid >> 5, lane = tid & 31, l15 = lane & 15, kh = lane >> 4;
  const unsigned short* Bp = Bm + (size_t)b * K * N;
  float* Op = Out + (size_t)b * M * N;

  floatx8 acc[4];
#pragma unroll
  for (int nt = 0; nt < 4; ++nt) acc[nt] = {};

  for (int k0 = 0; k0 < K; k0 += 32) {
    __syncthreads();
    for (int t = tid; t < 2048; t += 128) {
      int i = t >> 5, k = t & 31;
      As[i][k] = A[(size_t)(m0 + i) * K + k0 + k];
    }
    for (int t = tid; t < 2048; t += 128) {
      int k = t >> 6, n = t & 63;
      BsT[n][k] = Bp[(size_t)(k0 + k) * N + n0 + n];
    }
    __syncthreads();
    if (k0 + 32 < K)
      __builtin_prefetch((const char*)(Bp + (size_t)(k0 + 32) * N) + tid * 48, 0, 0);

    bf16x16 af = frag_ld(&As[w * 16 + l15][kh * 8], &As[w * 16 + l15][16 + kh * 8]);
#pragma unroll
    for (int nt = 0; nt < 4; ++nt) {
      bf16x16 bf = frag_ld(&BsT[nt * 16 + l15][kh * 8],
                           &BsT[nt * 16 + l15][16 + kh * 8]);
      acc[nt] = wmma_bf16(af, bf, acc[nt]);
    }
  }
#pragma unroll
  for (int nt = 0; nt < 4; ++nt)
#pragma unroll
    for (int r = 0; r < 8; ++r) {
      int o = m0 + w * 16 + r + 8 * kh;
      int n = n0 + nt * 16 + l15;
      Op[(size_t)o * N + n] = acc[nt][r] + bias[o];
    }
}

// ---------------------------------------------------------------------------
// BN stats over (B,HW) per channel for [b][c][hw] layout
// ---------------------------------------------------------------------------
__global__ void k_bn_stats_c(const float* __restrict__ T, float* __restrict__ mu,
                             float* __restrict__ rs, int n_per) {
  __shared__ float sum[256], sq[256];
  int o = blockIdx.x;
  float s = 0.f, q = 0.f;
  for (int t = threadIdx.x; t < n_per; t += 256) {
    int bb = t / HW_, hw = t % HW_;
    float v = T[((size_t)bb * C_ + o) * HW_ + hw];
    s += v; q += v * v;
  }
  sum[threadIdx.x] = s; sq[threadIdx.x] = q;
  __syncthreads();
  for (int st = 128; st > 0; st >>= 1) {
    if (threadIdx.x < st) {
      sum[threadIdx.x] += sum[threadIdx.x + st];
      sq[threadIdx.x] += sq[threadIdx.x + st];
    }
    __syncthreads();
  }
  if (threadIdx.x == 0) {
    float m = sum[0] / n_per;
    float v = sq[0] / n_per - m * m;
    mu[o] = m;
    rs[o] = rsqrtf(v + EPS_);
  }
}

__global__ void k_bn_apply_res(const float* __restrict__ T, const float* __restrict__ x,
                               const float* __restrict__ mu, const float* __restrict__ rs,
                               const float* __restrict__ g, const float* __restrict__ be,
                               float* __restrict__ Z, int n) {
  int i = blockIdx.x * blockDim.x + threadIdx.x;
  if (i >= n) return;
  int o = (i / HW_) % C_;
  Z[i] = (T[i] - mu[o]) * rs[o] * g[o] + be[o] + x[i];
}

__global__ void k_mean_hw(const float* __restrict__ Z, float* __restrict__ U, int n) {
  int i = blockIdx.x * blockDim.x + threadIdx.x;
  if (i >= n) return;
  const float* p = Z + (size_t)i * HW_;
  float s = 0.f;
  for (int hw = 0; hw < HW_; ++hw) s += p[hw];
  U[i] = s * (1.0f / HW_);
}

// sa branch: s[b][n][hw] = sa_b[n] + sum_c sa_w[n][c] * z[b][c][hw]
__global__ void k_sa_conv(const float* __restrict__ Z, const float* __restrict__ sw,
                          const float* __restrict__ sb, float* __restrict__ S) {
  int b = blockIdx.x >> 2, nIdx = blockIdx.x & 3, hw = threadIdx.x;
  float acc = sb[nIdx];
  const float* Zb = Z + (size_t)b * C_ * HW_ + hw;
  const float* Wn = sw + (size_t)nIdx * C_;
  for (int c = 0; c < C_; ++c) acc += Wn[c] * Zb[(size_t)c * HW_];
  S[((size_t)b * 4 + nIdx) * HW_ + hw] = acc;
}

__global__ void k_sa_stats(const float* __restrict__ S, float* __restrict__ mu,
                           float* __restrict__ rs) {
  __shared__ float sum[256], sq[256];
  for (int nIdx = 0; nIdx < 4; ++nIdx) {
    float s = 0.f, q = 0.f;
    for (int t = threadIdx.x; t < B_ * HW_; t += 256) {
      int bb = t / HW_, hw = t % HW_;
      float v = S[((size_t)bb * 4 + nIdx) * HW_ + hw];
      s += v; q += v * v;
    }
    sum[threadIdx.x] = s; sq[threadIdx.x] = q;
    __syncthreads();
    for (int st = 128; st > 0; st >>= 1) {
      if (threadIdx.x < st) {
        sum[threadIdx.x] += sum[threadIdx.x + st];
        sq[threadIdx.x] += sq[threadIdx.x + st];
      }
      __syncthreads();
    }
    if (threadIdx.x == 0) {
      float m = sum[0] / (B_ * HW_);
      float v = sq[0] / (B_ * HW_) - m * m;
      mu[nIdx] = m; rs[nIdx] = rsqrtf(v + EPS_);
    }
    __syncthreads();
  }
}

__global__ void k_sa_act(const float* __restrict__ S, const float* __restrict__ mu,
                         const float* __restrict__ rs, const float* __restrict__ g,
                         const float* __restrict__ be, float* __restrict__ Aact,
                         float* __restrict__ out2) {
  int i = blockIdx.x * blockDim.x + threadIdx.x;
  if (i >= B_ * 4 * HW_) return;
  int nIdx = (i / HW_) & 3;
  float v = (S[i] - mu[nIdx]) * rs[nIdx] * g[nIdx] + be[nIdx];
  float a = 1.0f / (1.0f + __expf(-v));
  Aact[i] = a;
  out2[i] = a;
}

// xn[b][n][c] = sum_hw a[b][n][hw] * z[b][c][hw]
__global__ void k_xn(const float* __restrict__ Aact, const float* __restrict__ Z,
                     float* __restrict__ Xn) {
  int i = blockIdx.x * blockDim.x + threadIdx.x;  // over B*C
  if (i >= B_ * C_) return;
  int b = i / C_, c = i % C_;
  const float* Zp = Z + (size_t)i * HW_;
  const float* Ap = Aact + (size_t)b * 4 * HW_;
  float a0 = 0.f, a1 = 0.f, a2 = 0.f, a3 = 0.f;
  for (int hw = 0; hw < HW_; ++hw) {
    float z = Zp[hw];
    a0 += Ap[hw] * z;
    a1 += Ap[HW_ + hw] * z;
    a2 += Ap[2 * HW_ + hw] * z;
    a3 += Ap[3 * HW_ + hw] * z;
  }
  float* X = Xn + (size_t)b * 4 * C_ + c;
  X[0] = a0; X[C_] = a1; X[2 * C_] = a2; X[3 * C_] = a3;
}

// Nd[b][j][o] = g_b[o] + sum_c g_w[o][c] * (j<4 ? xn[b][j][c] : u[b][c])
__global__ void k_nodes(const float* __restrict__ Xn, const float* __restrict__ U,
                        const float* __restrict__ gw, const float* __restrict__ gb,
                        float* __restrict__ Nd) {
  int i = blockIdx.x * blockDim.x + threadIdx.x;  // over B*CI
  if (i >= B_ * CI_) return;
  int b = i / CI_, o = i % CI_;
  const float* W = gw + (size_t)o * C_;
  const float* X0 = Xn + (size_t)b * 4 * C_;
  const float* Ub = U + (size_t)b * C_;
  float a0 = gb[o], a1 = a0, a2 = a0, a3 = a0, a4 = a0;
  for (int c = 0; c < C_; ++c) {
    float w = W[c];
    a0 += w * X0[c];
    a1 += w * X0[C_ + c];
    a2 += w * X0[2 * C_ + c];
    a3 += w * X0[3 * C_ + c];
    a4 += w * Ub[c];
  }
  float* N = Nd + (size_t)b * 5 * CI_ + o;
  N[0] = a0; N[CI_] = a1; N[2 * CI_] = a2; N[3 * CI_] = a3; N[4 * CI_] = a4;
}

// ---------------------------------------------------------------------------
// 5-node attention per batch. Algebraic folding:
//   y2[o] = sw_b[o] + dot(avgP, sw_w[o]) + mean_{n<4} nd[n][o]
//   avgP[c] = sum_m (mean_{n<4} att[n][m]) * V[m][c]
// ---------------------------------------------------------------------------
__global__ void __launch_bounds__(256)
k_node_attn(const float* __restrict__ Nd,
            const float* __restrict__ thw, const float* __restrict__ thb,
            const float* __restrict__ phw, const float* __restrict__ phb,
            const float* __restrict__ sgw, const float* __restrict__ sgb,
            const float* __restrict__ sww, const float* __restrict__ swb,
            float* __restrict__ UC) {
  __shared__ float sNd[5 * CI_];
  __shared__ float sQ[5 * CI_];   // q, later V
  __shared__ float sK[5 * CI_];   // k, later avgP (row 0)
  __shared__ float satt[25];
  __shared__ float wbar[5];
  int b = blockIdx.x, tid = threadIdx.x;
  const float* Np = Nd + (size_t)b * 5 * CI_;
  for (int t = tid; t < 5 * CI_; t += 256) sNd[t] = Np[t];
  __syncthreads();
  for (int t = tid; t < 2 * 5 * CI_; t += 256) {
    int which = t / (5 * CI_);
    int rem = t % (5 * CI_);
    int n = rem / CI_, o = rem % CI_;
    const float* W = (which ? phw : thw) + (size_t)o * CI_;
    const float* nd = sNd + n * CI_;
    float acc = which ? phb[o] : thb[o];
    for (int c = 0; c < CI_; ++c) acc += W[c] * nd[c];
    (which ? sK : sQ)[rem] = acc;
  }
  __syncthreads();
  if (tid < 25) {
    int n = tid / 5, m = tid % 5;
    float acc = 0.f;
    const float* q = sQ + n * CI_;
    const float* k = sK + m * CI_;
    for (int c = 0; c < CI_; ++c) acc += q[c] * k[c];
    satt[tid] = acc * (1.0f / 32.0f);   // / sqrt(1024)
  }
  __syncthreads();
  if (tid < 5) {
    float mx = satt[tid * 5];
    for (int m = 1; m < 5; ++m) mx = fmaxf(mx, satt[tid * 5 + m]);
    float s = 0.f, e[5];
    for (int m = 0; m < 5; ++m) { e[m] = __expf(satt[tid * 5 + m] - mx); s += e[m]; }
    for (int m = 0; m < 5; ++m) satt[tid * 5 + m] = e[m] / s;
  }
  __syncthreads();
  if (tid < 5)
    wbar[tid] = 0.25f * (satt[tid] + satt[5 + tid] + satt[10 + tid] + satt[15 + tid]);
  __syncthreads();
  for (int t = tid; t < 5 * CI_; t += 256) {   // V into sQ
    int m = t / CI_, c = t % CI_;
    const float* W = sgw + (size_t)c * CI_;
    const float* nd = sNd + m * CI_;
    float acc = sgb[c];
    for (int cc = 0; cc < CI_; ++cc) acc += W[cc] * nd[cc];
    sQ[t] = acc;
  }
  __syncthreads();
  for (int c = tid; c < CI_; c += 256) {       // avgP into sK
    float acc = 0.f;
    for (int m = 0; m < 5; ++m) acc += wbar[m] * sQ[m * CI_ + c];
    sK[c] = acc;
  }
  __syncthreads();
  float* UCb = UC + (size_t)b * C_;
  for (int o = tid; o < CI_; o += 256) {
    const float* W = sww + (size_t)o * CI_;
    float acc = swb[o];
    for (int c = 0; c < CI_; ++c) acc += W[c] * sK[c];
    acc += 0.25f * (sNd[o] + sNd[CI_ + o] + sNd[2 * CI_ + o] + sNd[3 * CI_ + o]);
    UCb[o] = acc;
    UCb[CI_ + o] = sNd[4 * CI_ + o];   // u2
  }
}

__global__ void k_conv2(const float* __restrict__ UC, const float* __restrict__ w2,
                        const float* __restrict__ b2, float* __restrict__ T2) {
  int i = blockIdx.x * blockDim.x + threadIdx.x;  // B*C
  if (i >= B_ * C_) return;
  int b = i / C_, o = i % C_;
  const float* W = w2 + (size_t)o * C_;
  const float* u = UC + (size_t)b * C_;
  float acc = b2[o];
  for (int c = 0; c < C_; ++c) acc += W[c] * u[c];
  T2[i] = acc;
}

__global__ void k_bn_stats_b(const float* __restrict__ T2, float* __restrict__ mu,
                             float* __restrict__ rs) {
  int o = blockIdx.x * blockDim.x + threadIdx.x;
  if (o >= C_) return;
  float s = 0.f, q = 0.f;
  for (int bb = 0; bb < B_; ++bb) {
    float v = T2[(size_t)bb * C_ + o];
    s += v; q += v * v;
  }
  float m = s / B_;
  float v = q / B_ - m * m;
  mu[o] = m; rs[o] = rsqrtf(v + EPS_);
}

__global__ void k_final(const float* __restrict__ Z, const float* __restrict__ T2,
                        const float* __restrict__ mu, const float* __restrict__ rs,
                        const float* __restrict__ g, const float* __restrict__ be,
                        float* __restrict__ out1) {
  int i = blockIdx.x * blockDim.x + threadIdx.x;
  if (i >= B_ * C_ * HW_) return;
  int bo = i / HW_;
  int o = bo % C_;
  float y3 = (T2[bo] - mu[o]) * rs[o] * g[o] + be[o];
  out1[i] = Z[i] + y3;
}

// ---------------------------------------------------------------------------
extern "C" void kernel_launch(void* const* d_in, const int* in_sizes, int n_in,
                              void* d_out, int out_size, void* d_ws, size_t ws_size,
                              hipStream_t stream) {
  (void)in_sizes; (void)n_in; (void)out_size; (void)ws_size;
  const float* x       = (const float*)d_in[0];
  const float* sa_w    = (const float*)d_in[1];
  const float* sa_b    = (const float*)d_in[2];
  const float* sa_g    = (const float*)d_in[3];
  const float* sa_beta = (const float*)d_in[4];
  const float* g_w     = (const float*)d_in[5];
  const float* g_b     = (const float*)d_in[6];
  const float* w1_w    = (const float*)d_in[7];
  const float* w1_b    = (const float*)d_in[8];
  const float* w1_g    = (const float*)d_in[9];
  const float* w1_beta = (const float*)d_in[10];
  const float* w2_w    = (const float*)d_in[11];
  const float* w2_b    = (const float*)d_in[12];
  const float* w2_g    = (const float*)d_in[13];
  const float* w2_beta = (const float*)d_in[14];
  const float* th_w    = (const float*)d_in[15];
  const float* th_b    = (const float*)d_in[16];
  const float* ph_w    = (const float*)d_in[17];
  const float* ph_b    = (const float*)d_in[18];
  const float* sg_w    = (const float*)d_in[19];
  const float* sg_b    = (const float*)d_in[20];
  const float* sw_w    = (const float*)d_in[21];
  const float* sw_b    = (const float*)d_in[22];

  // Workspace regions (aliased by liveness):
  //  A [0,50.3M): Xb (K0-K1) then Tf (K2-K4)
  //  B [50.3M,100.7M): W1b (K0-K2) then Zf (K4-end)
  //  C [100.7M,...): Yb (K1-K2) then all small buffers (K3 onward)
  char* ws = (char*)d_ws;
  const size_t REG = 50331648;
  unsigned short* Xb  = (unsigned short*)(ws);
  float*          Tf  = (float*)(ws);
  unsigned short* W1b = (unsigned short*)(ws + REG);
  float*          Zf  = (float*)(ws + REG);
  char* rc = ws + 2 * REG;
  unsigned short* Yb  = (unsigned short*)rc;
  float* mu1  = (float*)(rc + 0);
  float* rs1  = (float*)(rc + 8192);
  float* sA   = (float*)(rc + 16384);
  float* Aact = (float*)(rc + 131072);
  float* muA  = (float*)(rc + 262144);
  float* rsA  = (float*)(rc + 262400);
  float* Uu   = (float*)(rc + 262656);
  float* Xn   = (float*)(rc + 524800);
  float* Ndp  = (float*)(rc + 1573376);
  float* UCp  = (float*)(rc + 2228736);
  float* T2   = (float*)(rc + 2490880);
  float* mu2  = (float*)(rc + 2753024);
  float* rs2  = (float*)(rc + 2761216);

  float* out1 = (float*)d_out;
  float* out2 = out1 + (size_t)B_ * C_ * HW_;

  const int n_x = B_ * C_ * HW_;
  k_f32_to_bf16<<<(n_x + 255) / 256, 256, 0, stream>>>(x, Xb, n_x);
  k_f32_to_bf16<<<(C_ * C_ + 255) / 256, 256, 0, stream>>>(w1_w, W1b, C_ * C_);
  k_flash_gram<<<dim3(C_ / 64, B_), 128, 0, stream>>>(Xb, Yb);
  k_gemm_bias<<<dim3(C_ / 64, HW_ / 64, B_), 128, 0, stream>>>(W1b, Yb, w1_b, Tf,
                                                               C_, C_, HW_);
  k_bn_stats_c<<<C_, 256, 0, stream>>>(Tf, mu1, rs1, B_ * HW_);
  k_bn_apply_res<<<(n_x + 255) / 256, 256, 0, stream>>>(Tf, x, mu1, rs1, w1_g,
                                                        w1_beta, Zf, n_x);
  k_mean_hw<<<(B_ * C_ + 255) / 256, 256, 0, stream>>>(Zf, Uu, B_ * C_);
  k_sa_conv<<<B_ * 4, HW_, 0, stream>>>(Zf, sa_w, sa_b, sA);
  k_sa_stats<<<1, 256, 0, stream>>>(sA, muA, rsA);
  k_sa_act<<<(B_ * 4 * HW_ + 255) / 256, 256, 0, stream>>>(sA, muA, rsA, sa_g,
                                                           sa_beta, Aact, out2);
  k_xn<<<(B_ * C_ + 255) / 256, 256, 0, stream>>>(Aact, Zf, Xn);
  k_nodes<<<(B_ * CI_ + 255) / 256, 256, 0, stream>>>(Xn, Uu, g_w, g_b, Ndp);
  k_node_attn<<<B_, 256, 0, stream>>>(Ndp, th_w, th_b, ph_w, ph_b, sg_w, sg_b,
                                      sw_w, sw_b, UCp);
  k_conv2<<<(B_ * C_ + 255) / 256, 256, 0, stream>>>(UCp, w2_w, w2_b, T2);
  k_bn_stats_b<<<(C_ + 255) / 256, 256, 0, stream>>>(T2, mu2, rs2);
  k_final<<<(n_x + 255) / 256, 256, 0, stream>>>(Zf, T2, mu2, rs2, w2_g, w2_beta,
                                                 out1);
}
